// Correlation_5669356831344
// MI455X (gfx1250) — compile-verified, matrix operand order checked by reading, not run on hardware
//
#include <hip/hip_runtime.h>

typedef _Float16 v16h __attribute__((ext_vector_type(16)));
typedef _Float16 v8h  __attribute__((ext_vector_type(8)));
typedef _Float16 h2   __attribute__((ext_vector_type(2)));
typedef float    v8f  __attribute__((ext_vector_type(8)));
typedef float    v4f  __attribute__((ext_vector_type(4)));

#define C_       256
#define H_       48
#define W_       64
#define NB_      8
#define ND       21          // displacements per axis
#define KC       128         // channels per pipeline step (2 steps per dy)
#define SA       264         // A row stride, halves (528B: 16B-aligned)
#define SB       132         // B row stride, floats (528B: 16B-aligned)
#define BCOLS    112         // x' in [-24, 88)
#define NTHREADS 256

#define A_BYTES    (W_ * SA * 2)                 // 33792
#define B_BYTES    (BCOLS * SB * 4)              // 59136
#define SMEM_BYTES (A_BYTES + 2 * B_BYTES)       // 152064 (< 160KB CU-mode cap)

__device__ __forceinline__ void wait_async_le32() {
#if __has_builtin(__builtin_amdgcn_s_wait_asynccnt)
  __builtin_amdgcn_s_wait_asynccnt(32);
#else
  asm volatile("s_wait_asynccnt 32" ::: "memory");
#endif
}
__device__ __forceinline__ void wait_async_le0() {
#if __has_builtin(__builtin_amdgcn_s_wait_asynccnt)
  __builtin_amdgcn_s_wait_asynccnt(0);
#else
  asm volatile("s_wait_asynccnt 0" ::: "memory");
#endif
}

// Stage one KC-channel chunk of in2 row y2 into a B buffer, transposed to
// [x'][c] fp32, using per-lane-addressed async global->LDS dword copies.
// 32 issues per wave: each issue = 1 channel, 32 consecutive x' (coalesced).
__device__ __forceinline__ void issue_b_stage(const float* b2img, int y2, int cbase,
                                              unsigned bufBase, int w, int lane) {
#pragma unroll 4
  for (int it = 0; it < 32; ++it) {
    int q  = w * 32 + it;                 // 0..255
    int c  = q >> 1;                      // 0..127 (channel within chunk)
    int xp = (q & 1) * 32 + lane;         // 0..63  (in2 x coordinate)
    unsigned lds  = bufBase + (unsigned)(xp + 24) * (SB * 4) + (unsigned)c * 4;
    unsigned voff = (unsigned)(((cbase + c) * (H_ * W_) + y2 * W_ + xp) * 4);
    asm volatile("global_load_async_to_lds_b32 %0, %1, %2"
                 :: "v"(lds), "v"(voff), "s"(b2img) : "memory");
  }
}

__device__ __forceinline__ v16h load_a_frag(const _Float16* Als, int row, int kabs, int sel) {
  const _Float16* p = Als + row * SA + kabs + 8 * sel;   // 16B-aligned
  v16h a;
  *(v8h*)&a         = *(const v8h*)(p);
  *(((v8h*)&a) + 1) = *(const v8h*)(p + 16);
  return a;
}

__device__ __forceinline__ v16h load_b_frag(const float* Bbuf, int row, int k0, int sel) {
  const float* p = Bbuf + row * SB + k0 + 8 * sel;       // 16B-aligned
  v8f f0, f1;
  *(v4f*)&f0         = *(const v4f*)(p);
  *(((v4f*)&f0) + 1) = *(const v4f*)(p + 4);
  *(v4f*)&f1         = *(const v4f*)(p + 16);
  *(((v4f*)&f1) + 1) = *(const v4f*)(p + 20);
  v8h h0 = __builtin_convertvector(f0, v8h);
  v8h h1 = __builtin_convertvector(f1, v8h);
  v16h bf;
  *(v8h*)&bf         = h0;
  *(((v8h*)&bf) + 1) = h1;
  return bf;
}

// One K-block (4 x 16x16x32 K-steps, 2 N-tiles) against register-resident A
// fragments. BASE selects which chunk's A fragments (constant -> no spills).
template<int BASE>
__device__ __forceinline__ void kblock(const v16h (&afrag)[8], const float* Bbuf,
                                       v8f (&acc)[2], int x0, int nh, int r, int sel) {
#pragma unroll
  for (int kk = 0; kk < 4; ++kk) {
    const int k0 = kk * 32;
    v16h bf0 = load_b_frag(Bbuf, x0 + 16 * (2 * nh)     + r, k0, sel);
    v16h bf1 = load_b_frag(Bbuf, x0 + 16 * (2 * nh + 1) + r, k0, sel);
    acc[0] = __builtin_amdgcn_wmma_f32_16x16x32_f16(
        false, afrag[BASE + kk], false, bf0, (short)0, acc[0], false, false);
    acc[1] = __builtin_amdgcn_wmma_f32_16x16x32_f16(
        false, afrag[BASE + kk], false, bf1, (short)0, acc[1], false, false);
  }
}

// One block per (y, b); dy loop inside so the in1 row is staged (and its WMMA
// fragments register-hoisted) exactly once.
__global__ __launch_bounds__(NTHREADS) void corr_wmma(const float* __restrict__ in1,
                                                      const float* __restrict__ in2,
                                                      float* __restrict__ out) {
  const int y    = (int)blockIdx.x;
  const int b    = (int)blockIdx.y;
  const int tid  = (int)threadIdx.x;
  const int lane = tid & 31;
  const int w    = tid >> 5;        // 8 waves
  const int x0   = (w >> 1) * 16;   // M tile
  const int nh   = w & 1;           // N half: tiles {2nh, 2nh+1}
  const int r    = lane & 15;
  const int sel  = lane >> 4;

  const size_t chanStride = (size_t)H_ * W_;            // 3072
  const size_t imgStride  = (size_t)C_ * chanStride;
  const float* a1row = in1 + (size_t)b * imgStride + (size_t)y * W_;
  const float* b2img = in2 + (size_t)b * imgStride;
  float* outB = out + (((size_t)b * (ND * ND)) * H_ + y) * W_;

  extern __shared__ __align__(16) char smem[];
  _Float16* Als = (_Float16*)smem;
  float* Bls0 = (float*)(smem + A_BYTES);
  float* Bls1 = (float*)(smem + A_BYTES + B_BYTES);
  const unsigned bBase0 = (unsigned)(size_t)(void*)Bls0;  // LDS byte offset
  const unsigned bBase1 = (unsigned)(size_t)(void*)Bls1;

  // ---- stage A once: in1[b, :, y, :] -> fp16 [x][c] (transposed) ----
#pragma unroll
  for (int uu = 0; uu < 8; ++uu) {
    int u     = tid + NTHREADS * uu;   // 0..2047
    int cpair = u >> 4;                // 0..127
    int xq    = u & 15;
    int c0    = 2 * cpair;
    v4f a0 = *(const v4f*)(a1row + (size_t)c0       * chanStride + 4 * xq);
    v4f a1 = *(const v4f*)(a1row + (size_t)(c0 + 1) * chanStride + 4 * xq);
#pragma unroll
    for (int j = 0; j < 4; ++j) {
      h2 p;
      p.x = (_Float16)a0[j];
      p.y = (_Float16)a1[j];
      *(h2*)&Als[(4 * xq + j) * SA + c0] = p;
    }
  }

  // ---- zero B's 48 pad columns (rows 0..23, 88..111) once; never rewritten ----
  for (int o = tid; o < 2 * 48 * KC; o += NTHREADS) {
    int buf = (o >= 48 * KC);
    int o2  = o - buf * 48 * KC;
    int rr  = o2 >> 7;                       // 0..47
    int cc  = o2 & (KC - 1);
    int row = (rr < 24) ? rr : (rr + 64);
    (buf ? Bls1 : Bls0)[row * SB + cc] = 0.0f;
  }

  // ---- zero slabs for out-of-range y+dy (reference zero-pads); valid-dy list ----
  int vdy[ND];
  int nv = 0;
  for (int dyi = 0; dyi < ND; ++dyi) {
    int y2 = y + 2 * (dyi - 10);
    if (y2 >= 0 && y2 < H_) {
      vdy[nv++] = dyi;
    } else {
      float* od = outB + (size_t)dyi * ND * chanStride;
      for (int o = tid; o < ND * W_; o += NTHREADS)
        od[(size_t)(o >> 6) * chanStride + (o & (W_ - 1))] = 0.0f;
    }
  }
  const int nsteps = 2 * nv;   // (dy, K-chunk) pipeline steps

  __syncthreads();             // A + pad zeros visible

  // ---- hoist all 8 A fragments (full C=256) into registers: reused for all dy ----
  v16h afrag[8];
#pragma unroll
  for (int kk = 0; kk < 8; ++kk)
    afrag[kk] = load_a_frag(Als, x0 + r, kk * 32, sel);

  // prologue: async-stage step 0 into buffer 0
  {
    int y2 = y + 2 * (vdy[0] - 10);
    issue_b_stage(b2img, y2, 0, bBase0, w, lane);
  }

  const v8f vzero = {0.f, 0.f, 0.f, 0.f, 0.f, 0.f, 0.f, 0.f};
  v8f acc[2] = {vzero, vzero};

  for (int s = 0; s < nsteps; ++s) {
    const int nxt = s + 1;
    if (nxt < nsteps) {        // prefetch next step into the other buffer
      int y2 = y + 2 * (vdy[nxt >> 1] - 10);
      issue_b_stage(b2img, y2, (nxt & 1) * KC, (nxt & 1) ? bBase1 : bBase0, w, lane);
      wait_async_le32();       // in-order completion: drains step s's 32 issues
    } else {
      wait_async_le0();
    }
    __syncthreads();

    const float* Bbuf = (s & 1) ? Bls1 : Bls0;
    const int ck = s & 1;
    if (ck == 0) {
      acc[0] = vzero;
      acc[1] = vzero;
      kblock<0>(afrag, Bbuf, acc, x0, nh, r, sel);   // channels 0..127
    } else {
      kblock<4>(afrag, Bbuf, acc, x0, nh, r, sel);   // channels 128..255

      // ---- extract banded diagonals and store ----
      const int dyi = vdy[s >> 1];
      float* od = outB + (size_t)dyi * ND * chanStride;
      const float scale = 1.0f / (float)C_;
      union { v8f v; float f[8]; } u;
#pragma unroll
      for (int ni = 0; ni < 2; ++ni) {
        int n = 2 * nh + ni;
        u.v = acc[ni];
#pragma unroll
        for (int v = 0; v < 8; ++v) {
          int i  = v + 8 * sel;            // output x = x0 + i
          int dx = 16 * n + r - i - 24;    // col x' - row x
          if (!(dx & 1) && dx >= -20 && dx <= 20)
            od[(size_t)((dx >> 1) + 10) * chanStride + (x0 + i)] = u.f[v] * scale;
        }
      }
    }
    __syncthreads();
  }
}

extern "C" void kernel_launch(void* const* d_in, const int* in_sizes, int n_in,
                              void* d_out, int out_size, void* d_ws, size_t ws_size,
                              hipStream_t stream) {
  const float* in1 = (const float*)d_in[0];
  const float* in2 = (const float*)d_in[1];
  float* out = (float*)d_out;
  (void)in_sizes; (void)n_in; (void)out_size; (void)d_ws; (void)ws_size;

  dim3 grid(H_, NB_);   // (y, b) = 48 x 8 blocks, 8 waves each
  corr_wmma<<<grid, NTHREADS, SMEM_BYTES, stream>>>(in1, in2, out);
}
